// SphericalVAE_16509854286076
// MI455X (gfx1250) — compile-verified, matrix operand order checked by reading, not run on hardware
//
#include <hip/hip_runtime.h>

typedef __attribute__((ext_vector_type(16))) _Float16 v16h;
typedef __attribute__((ext_vector_type(8)))  float    v8f;
typedef __attribute__((ext_vector_type(4)))  unsigned v4u;
typedef __attribute__((ext_vector_type(4)))  int      v4i;
typedef __attribute__((ext_vector_type(8)))  int      v8i;

#define B_   4
#define C_   32
#define CO_  32
#define N_   32768
#define K_   17
#define J_   16
#define WELEMS (CO_ * C_ * J_)          // 16384 weight elements
#define TILES_PER_B (N_ / 16)           // 2048
#define NTILES      (B_ * TILES_PER_B)  // 8192
#define LDS_BYTES   (WELEMS * 4 + WELEMS * 2) // 64KB f32 raw + 32KB f16 swizzled

__global__ void zero_out_kernel(float* __restrict__ out, int n) {
  int i = blockIdx.x * blockDim.x + threadIdx.x;
  int stride = gridDim.x * blockDim.x;
  for (; i < n; i += stride) out[i] = 0.0f;
}

__global__ __launch_bounds__(256) void edgeconv_wmma_kernel(
    const float* __restrict__ x,     // [B, C, N]
    const int*   __restrict__ adj,   // [B, N, K]
    const float* __restrict__ w,     // [CO, C, J]
    float*       __restrict__ out) { // [B, CO, N]
  extern __shared__ char smem[];
  float*    wRaw  = (float*)smem;                 // 64KB, memory order [o][c][j]
  _Float16* wFrag = (_Float16*)(smem + WELEMS * 4); // 32KB, [j][tile][lane][16]

  // ---- Stage 1: TDM async copy of the whole weight tensor into LDS ----
#if __has_builtin(__builtin_amdgcn_tensor_load_to_lds)
  if ((threadIdx.x >> 5) == 0) { // one wave issues the DMA (EXEC ignored by TDM)
    unsigned long long ga = (unsigned long long)(uintptr_t)w;
    unsigned ldsa = (unsigned)(uintptr_t)wRaw; // low 32b of flat addr = LDS offset
    // D# group0: count=1 | lds_addr | global_addr[56:0] | type=2
    v4u g0 = { 1u,
               ldsa,
               (unsigned)ga,
               ((unsigned)((ga >> 32) & 0x01FFFFFFull)) | (2u << 30) };
    // D# group1: data_size=2(4B); tensor_dim0=16384; tensor_dim1=1;
    //            tile_dim0=16384; tile_dim1/2 unused; dim0_stride=16384
    v8i g1 = { 0x20000,     // [17:16]=data_size=2, mask/pad/iterate = 0
               0x40000000,  // tensor_dim0 low16 (=16384) in bits[63:48]
               0x00010000,  // tensor_dim0 hi=0, tensor_dim1=1 (low16)
               0x40000000,  // tensor_dim1 hi=0, tile_dim0=16384 in [127:112]
               0,           // tile_dim1=0, tile_dim2=0 (unused)
               16384, 0,    // tensor_dim0_stride = 16384 (48-bit)
               0 };         // tensor_dim1_stride = 0
    v4i gz = { 0, 0, 0, 0 };
#if defined(__clang_major__) && (__clang_major__ >= 23)
    v8i gz8 = { 0, 0, 0, 0, 0, 0, 0, 0 };
    __builtin_amdgcn_tensor_load_to_lds(g0, g1, gz, gz, gz8, 0);
#else
    __builtin_amdgcn_tensor_load_to_lds(g0, g1, gz, gz, 0);
#endif
    __builtin_amdgcn_s_wait_tensorcnt(0);
  }
#else
  for (int e = threadIdx.x; e < WELEMS; e += blockDim.x) wRaw[e] = w[e];
#endif
  __syncthreads();

  // ---- Stage 2: convert f32 -> f16 into per-lane-contiguous WMMA A layout ----
  // wFrag flat index f = ((j*2 + tile)*32 + lane)*16 + i
  for (int f = threadIdx.x; f < WELEMS; f += blockDim.x) {
    int i    = f & 15;
    int lane = (f >> 4) & 31;
    int t    = (f >> 9) & 1;
    int j    = f >> 10;
    int hl2  = lane >> 4;
    int row  = (lane & 15) + 16 * t;                       // out channel o
    int r    = i >> 1;
    int k    = (r < 4 ? 2 * r : 16 + 2 * (r - 4)) + 8 * hl2 + (i & 1);
    wFrag[f] = (_Float16)wRaw[(row * C_ + k) * J_ + j];
  }
  __syncthreads();

  // ---- Stage 3: per-wave 16-point tile, loop over 16 neighbors ----
  const int lane = threadIdx.x & 31;
  const int hl   = lane >> 4;  // K-split half of the wave
  const int col  = lane & 15;  // point within tile / D column

  const int waveId = blockIdx.x * (blockDim.x >> 5) + (threadIdx.x >> 5);
  if (waveId >= NTILES) return; // wave-uniform (never taken at this grid size)
  const int b  = waveId / TILES_PER_B;
  const int n0 = (waveId % TILES_PER_B) * 16;
  const int n  = n0 + col;

  const float* xb     = x + (size_t)b * C_ * N_;
  const int*   adjRow = adj + ((size_t)b * N_ + n) * K_ + 1; // knn_idx = adj[:,:,1:]
  float*       outb   = out + (size_t)b * CO_ * N_;

  // Channel (K) mapping for this lane, matching the f16 WMMA A/B layout:
  // lanes 0-15 own channels {0..7, 16..23}; lanes 16-31 own {8..15, 24..31}.
  int kmap[8];
#pragma unroll
  for (int r = 0; r < 8; ++r)
    kmap[r] = (r < 4 ? 2 * r : 16 + 2 * (r - 4)) + 8 * hl;

  // Cache this lane's center-point channels (reused across all 16 neighbors).
  const float* xcen = xb + n;
  float xc[16];
#pragma unroll
  for (int r = 0; r < 8; ++r) {
    xc[2 * r]     = xcen[(size_t)kmap[r] * N_];
    xc[2 * r + 1] = xcen[(size_t)(kmap[r] + 1) * N_];
  }

  const float inv_k = 1.0f / (float)K_;
  const _Float16* pa = &wFrag[lane * 16]; // this lane's fragment, j-stride 1024

  int idx = adjRow[0];
  for (int j = 0; j < J_; ++j) {
    const int idx_next = (j + 1 < J_) ? adjRow[j + 1] : idx;
    const float* xnbN = xb + idx_next;
    __builtin_prefetch(&xnbN[(size_t)kmap[0] * N_], 0, 0); // global_prefetch_b8
    __builtin_prefetch(&xnbN[(size_t)kmap[4] * N_], 0, 0);

    // Gather neighbor column, edge diff, half-wave partial squared norm.
    const float* xnb = xb + idx;
    float diff[16];
    float ss = 0.0f;
#pragma unroll
    for (int r = 0; r < 8; ++r) {
      float d0 = xnb[(size_t)kmap[r] * N_]       - xc[2 * r];
      float d1 = xnb[(size_t)(kmap[r] + 1) * N_] - xc[2 * r + 1];
      diff[2 * r]     = d0;
      diff[2 * r + 1] = d1;
      ss += d0 * d0 + d1 * d1;
    }
    float nrm2  = ss + __shfl_xor(ss, 16);   // full 32-channel ||diff||^2
    float scale = rsqrtf(2.0f * nrm2);       // proj = diff * rsqrt(2*||diff||^2)

    v16h bfrag;
#pragma unroll
    for (int i = 0; i < 16; ++i) bfrag[i] = (_Float16)(diff[i] * scale);

    // A fragments: one 32B contiguous LDS read per tile (2x ds_load_b128 each).
    v16h a0 = *(const v16h*)(pa + (size_t)(j * 2 + 0) * 512);
    v16h a1 = *(const v16h*)(pa + (size_t)(j * 2 + 1) * 512);

    v8f z = {0.f, 0.f, 0.f, 0.f, 0.f, 0.f, 0.f, 0.f};
    // (neg_a, A, neg_b, B, c_mod, C, reuse_a, reuse_b)
    v8f d0 = __builtin_amdgcn_wmma_f32_16x16x32_f16(false, a0, false, bfrag,
                                                    (short)0, z, false, false);
    v8f d1 = __builtin_amdgcn_wmma_f32_16x16x32_f16(false, a1, false, bfrag,
                                                    (short)0, z, false, false);

    // Scatter: D column N = lane&15 -> this lane's idx; row o = r + 8*hl.
    float* po = outb + idx; // constant byte offsets below fit 24-bit IOFFSET
#pragma unroll
    for (int r = 0; r < 8; ++r) {
      int o = r + 8 * hl;
      atomicAdd(&po[(size_t)o * N_],        d0[r] * inv_k);
      atomicAdd(&po[(size_t)(o + 16) * N_], d1[r] * inv_k);
    }
    idx = idx_next;
  }
}

extern "C" void kernel_launch(void* const* d_in, const int* in_sizes, int n_in,
                              void* d_out, int out_size, void* d_ws, size_t ws_size,
                              hipStream_t stream) {
  const float* x   = (const float*)d_in[0];
  const int*   adj = (const int*)d_in[1];
  const float* w   = (const float*)d_in[2];
  float*       out = (float*)d_out;

  // Harness poisons d_out: zero it every call (atomically accumulated below).
  zero_out_kernel<<<512, 256, 0, stream>>>(out, B_ * CO_ * N_);

  // 8192 tiles, one wave32 per tile, 8 waves (256 threads) per block.
  edgeconv_wmma_kernel<<<NTILES / 8, 256, LDS_BYTES, stream>>>(x, adj, w, out);
}
